// GATv2_71433896067129
// MI455X (gfx1250) — compile-verified, matrix operand order checked by reading, not run on hardware
//
#include <hip/hip_runtime.h>
#include <hip/hip_bf16.h>
#include <cstdint>

typedef _Float16 half_t;
typedef __attribute__((ext_vector_type(16))) _Float16 v16h;
typedef __attribute__((ext_vector_type(8)))  _Float16 v8h;
typedef __attribute__((ext_vector_type(2)))  _Float16 v2h;
typedef __attribute__((ext_vector_type(8)))  float    v8f;

#define NEG_SLOPE 0.2f

__device__ __forceinline__ unsigned fenc(float f) {
    unsigned u = __float_as_uint(f);
    return (u & 0x80000000u) ? ~u : (u | 0x80000000u);
}
__device__ __forceinline__ float fdec(unsigned e) {
    return (e & 0x80000000u) ? __uint_as_float(e ^ 0x80000000u)
                             : __uint_as_float(~e);
}

// vectorized f16 -> f32 gather helper (PER = channels per lane)
template <int PER>
__device__ __forceinline__ void load_h(const half_t* __restrict__ p, float* f) {
    if constexpr (PER == 8) {
        v8h v = *(const v8h*)p;               // 16-B aligned -> b128
#pragma unroll
        for (int j = 0; j < 8; ++j) f[j] = (float)v[j];
    } else {
        v2h v = *(const v2h*)p;               // 4-B aligned -> b32
        f[0] = (float)v[0];
        f[1] = (float)v[1];
    }
}

// ---------------- conversions ----------------
__global__ __launch_bounds__(256)
void k_f32_to_f16(const float* __restrict__ in, half_t* __restrict__ out, size_t n) {
    size_t i = (size_t)blockIdx.x * blockDim.x + threadIdx.x;
    if (i < n) out[i] = (half_t)in[i];
}

// W is [K, Ncols] row-major f32 -> WT is [Ncols, K] row-major f16
__global__ __launch_bounds__(256)
void k_transpose_to_f16(const float* __restrict__ W, half_t* __restrict__ WT,
                        int K, int Ncols) {
    int i = blockIdx.x * blockDim.x + threadIdx.x;
    if (i < K * Ncols) {
        int k = i / Ncols, n = i % Ncols;
        WT[(size_t)n * K + k] = (half_t)W[i];
    }
}

__global__ __launch_bounds__(256)
void k_fill_u32(unsigned* __restrict__ p, unsigned v, size_t n) {
    size_t i = (size_t)blockIdx.x * blockDim.x + threadIdx.x;
    if (i < n) p[i] = v;
}

// ---------------- WMMA GEMM:  out[M,Ncols](f16) = X[M,K](f16) * WT[Ncols,K]^T + bias
// One wave computes one 16x16 output tile; K-loop of v_wmma_f32_16x16x32_f16.
__global__ __launch_bounds__(256)
void k_wmma_gemm_bias(const half_t* __restrict__ X, const half_t* __restrict__ WT,
                      const float* __restrict__ bias, half_t* __restrict__ out,
                      int M, int K, int Ncols) {
    const int wave   = blockIdx.x * (blockDim.x >> 5) + (threadIdx.x >> 5);
    const int ntiles = Ncols >> 4;
    const int mtile  = wave / ntiles;
    const int ntile  = wave - mtile * ntiles;
    if (mtile * 16 >= M) return;              // wave-uniform

    const int lane = threadIdx.x & 31;
    const int mrow = mtile * 16 + (lane & 15);
    const int ncol = ntile * 16 + (lane & 15);
    const int khi  = lane >> 4;               // 0 or 1
    // ISA 16-bit A layout: lanes 0-15: K 0-7 (V0-3) & 16-23 (V4-7); lanes 16-31: +8
    const half_t* arow = X  + (size_t)mrow * K + khi * 8;
    const half_t* brow = WT + (size_t)ncol * K + khi * 8;

    v8f c = {};
    const int ksteps = K >> 5;
    for (int kk = 0; kk < ksteps; ++kk) {
        v8h a_lo = *(const v8h*)(arow + kk * 32);
        v8h a_hi = *(const v8h*)(arow + kk * 32 + 16);
        v8h b_lo = *(const v8h*)(brow + kk * 32);
        v8h b_hi = *(const v8h*)(brow + kk * 32 + 16);
        v16h a, b;
#pragma unroll
        for (int i = 0; i < 8; ++i) {
            a[i] = a_lo[i]; a[i + 8] = a_hi[i];
            b[i] = b_lo[i]; b[i + 8] = b_hi[i];
        }
        c = __builtin_amdgcn_wmma_f32_16x16x32_f16(
                false, a, false, b, (short)0, c, false, false);
    }
    const float bv = bias[ncol];
    const int rbase = mtile * 16 + khi * 8;   // C layout: lanes 16-31 hold M=8..15
#pragma unroll
    for (int r = 0; r < 8; ++r) {
        int row = rbase + r;
        if (row < M) out[(size_t)row * Ncols + ncol] = (half_t)(c[r] + bv);
    }
}

// ---------------- edge sweep 1: score + segment max ----------------
// wave per edge; lane covers HC/32 contiguous channels (stays inside one head).
template <int H, int C>
__global__ __launch_bounds__(256)
void k_edge_score(const half_t* __restrict__ xs, const half_t* __restrict__ xd,
                  const float* __restrict__ att,
                  const int* __restrict__ srcA, const int* __restrict__ dstA,
                  int E0, int Etot, float* __restrict__ score,
                  unsigned* __restrict__ maxEnc) {
    int e = blockIdx.x * (blockDim.x >> 5) + (threadIdx.x >> 5);
    if (e >= Etot) return;
    const int lane = threadIdx.x & 31;
    int s, d;
    if (e < E0) { s = srcA[e]; d = dstA[e]; } else { s = e - E0; d = s; }
    constexpr int HC  = H * C;
    constexpr int per = HC / 32;      // channels per lane
    constexpr int lph = C / per;      // lanes per head
    const int ch0 = lane * per;
    float fs[per], fd[per];
    load_h<per>(xs + (size_t)s * HC + ch0, fs);
    load_h<per>(xd + (size_t)d * HC + ch0, fd);
    float partial = 0.f;
#pragma unroll
    for (int j = 0; j < per; ++j) {
        float v = fs[j] + fd[j];
        v = v > 0.f ? v : NEG_SLOPE * v;            // GATv2: lrelu before att dot
        partial += att[ch0 + j] * v;
    }
#pragma unroll
    for (int off = lph >> 1; off > 0; off >>= 1)
        partial += __shfl_xor(partial, off, 32);
    if ((lane % lph) == 0) {
        int h = lane / lph;
        score[(size_t)e * H + h] = partial;
        atomicMax(&maxEnc[(size_t)d * H + h], fenc(partial));
    }
}

// ---------------- edge sweep 2: exp + segment sum ----------------
template <int H>
__global__ __launch_bounds__(256)
void k_edge_ex(float* __restrict__ score, const int* __restrict__ dstA,
               int E0, int Etot, const unsigned* __restrict__ maxEnc,
               float* __restrict__ denom) {
    size_t idx = (size_t)blockIdx.x * blockDim.x + threadIdx.x;
    int e = (int)(idx / H), h = (int)(idx % H);
    if (e >= Etot) return;
    int d = (e < E0) ? dstA[e] : (e - E0);
    float m  = fdec(maxEnc[(size_t)d * H + h]);
    float ex = __expf(score[idx] - m);
    score[idx] = ex;                                  // in-place: score -> ex
    atomicAdd(&denom[(size_t)d * H + h], ex);
}

// ---------------- edge sweep 3: alpha-weighted scatter ----------------
template <int H, int C>
__global__ __launch_bounds__(256)
void k_edge_scatter(const half_t* __restrict__ xs, const float* __restrict__ ex,
                    const float* __restrict__ denom,
                    const int* __restrict__ srcA, const int* __restrict__ dstA,
                    int E0, int Etot, float* __restrict__ out) {
    int e = blockIdx.x * (blockDim.x >> 5) + (threadIdx.x >> 5);
    if (e >= Etot) return;
    const int lane = threadIdx.x & 31;
    int s, d;
    if (e < E0) { s = srcA[e]; d = dstA[e]; } else { s = e - E0; d = s; }
    constexpr int HC  = H * C;
    constexpr int per = HC / 32;
    constexpr int lph = C / per;
    const int ch0 = lane * per;
    const int h   = lane / lph;
    float alpha = ex[(size_t)e * H + h] / denom[(size_t)d * H + h];
    float fs[per];
    load_h<per>(xs + (size_t)s * HC + ch0, fs);
    float* po = out + (size_t)d * HC + ch0;
#pragma unroll
    for (int j = 0; j < per; ++j)
        atomicAdd(&po[j], alpha * fs[j]);
}

// ---------------- bias + relu + convert to f16 (feeds layer-2 GEMM) ------------
__global__ __launch_bounds__(256)
void k_bias_relu_f16(const float* __restrict__ acc, const float* __restrict__ bias,
                     half_t* __restrict__ out, int HC, size_t total) {
    size_t i = (size_t)blockIdx.x * blockDim.x + threadIdx.x;
    if (i < total) {
        float v = acc[i] + bias[(int)(i % HC)];
        out[i] = (half_t)(v > 0.f ? v : 0.f);
    }
}

// ---------------- mean-pool accumulation (bias2 fused — pooling is linear) -----
__global__ __launch_bounds__(256)
void k_pool_accum(const float* __restrict__ out2, const float* __restrict__ bias2,
                  const int* __restrict__ batch, float* __restrict__ sums,
                  float* __restrict__ counts, int N, int C2) {
    size_t i = (size_t)blockIdx.x * blockDim.x + threadIdx.x;
    if (i < (size_t)N * C2) {
        int n = (int)(i / C2), c = (int)(i % C2);
        int g = batch[n];
        atomicAdd(&sums[(size_t)g * C2 + c], out2[i] + bias2[c]);
        if (c == 0) atomicAdd(&counts[g], 1.0f);
    }
}

// ---------------- final: pooled @ W_lin + b_lin ----------------
__global__ __launch_bounds__(64)
void k_final_lin(const float* __restrict__ sums, const float* __restrict__ counts,
                 const float* __restrict__ Wlin, const float* __restrict__ blin,
                 float* __restrict__ out, int C2) {
    __shared__ float red[64];
    int g = blockIdx.x, c = threadIdx.x;
    float cnt = counts[g]; cnt = cnt > 1.f ? cnt : 1.f;
    red[c] = (sums[(size_t)g * C2 + c] / cnt) * Wlin[c];
    __syncthreads();
    for (int off = 32; off > 0; off >>= 1) {
        if (c < off) red[c] += red[c + off];
        __syncthreads();
    }
    if (c == 0) out[g] = red[0] + blin[0];
}

// =====================================================================
extern "C" void kernel_launch(void* const* d_in, const int* in_sizes, int n_in,
                              void* d_out, int out_size, void* d_ws, size_t ws_size,
                              hipStream_t stream) {
    const float* x     = (const float*)d_in[0];
    const int*   eidx  = (const int*)  d_in[1];
    const int*   batch = (const int*)  d_in[2];
    const float* W_l1  = (const float*)d_in[3];
    const float* b_l1  = (const float*)d_in[4];
    const float* W_r1  = (const float*)d_in[5];
    const float* b_r1  = (const float*)d_in[6];
    const float* att1  = (const float*)d_in[7];
    const float* bias1 = (const float*)d_in[8];
    const float* W_l2  = (const float*)d_in[9];
    const float* b_l2  = (const float*)d_in[10];
    const float* W_r2  = (const float*)d_in[11];
    const float* b_r2  = (const float*)d_in[12];
    const float* att2  = (const float*)d_in[13];
    const float* bias2 = (const float*)d_in[14];
    const float* W_lin = (const float*)d_in[15];
    const float* b_lin = (const float*)d_in[16];
    float* out = (float*)d_out;

    const int F = 128, HC1 = 256, C2 = 64;
    const int N    = in_sizes[0] / F;
    const int E    = in_sizes[1] / 2;
    const int Etot = E + N;                 // self-loops appended
    const int* srcA = eidx;
    const int* dstA = eidx + E;

    // ---- workspace carve-out (256-B aligned) ----
    size_t off = 0;
    auto alloc = [&](size_t bytes) -> void* {
        off = (off + 255) & ~(size_t)255;
        void* p = (char*)d_ws + off;
        off += bytes;
        return p;
    };
    half_t*  x16    = (half_t*) alloc((size_t)N * F * 2);
    half_t*  WlT1   = (half_t*) alloc((size_t)HC1 * F * 2);
    half_t*  WrT1   = (half_t*) alloc((size_t)HC1 * F * 2);
    half_t*  WlT2   = (half_t*) alloc((size_t)C2 * HC1 * 2);
    half_t*  WrT2   = (half_t*) alloc((size_t)C2 * HC1 * 2);
    half_t*  xs1    = (half_t*) alloc((size_t)N * HC1 * 2);   // f16: halves gather BW
    half_t*  xd1    = (half_t*) alloc((size_t)N * HC1 * 2);
    float*   out1   = (float*)  alloc((size_t)N * HC1 * 4);   // f32 atomic accum
    half_t*  h16    = (half_t*) alloc((size_t)N * HC1 * 2);
    float*   score1 = (float*)  alloc((size_t)Etot * 4 * 4);
    unsigned* max1  = (unsigned*)alloc((size_t)N * 4 * 4);
    float*   den1   = (float*)  alloc((size_t)N * 4 * 4);
    half_t*  xs2    = (half_t*) alloc((size_t)N * C2 * 2);
    half_t*  xd2    = (half_t*) alloc((size_t)N * C2 * 2);
    float*   out2   = (float*)  alloc((size_t)N * C2 * 4);
    float*   score2 = (float*)  alloc((size_t)Etot * 4);
    unsigned* max2  = (unsigned*)alloc((size_t)N * 4);
    float*   den2   = (float*)  alloc((size_t)N * 4);
    float*   sums   = (float*)  alloc((size_t)64 * C2 * 4);
    float*   counts = (float*)  alloc((size_t)64 * 4);

    auto cdiv = [](long a, long b) { return (int)((a + b - 1) / b); };
    const unsigned NEG_MAX_ENC = 0x00800000u;   // fenc(-FLT_MAX)

    // ---- convert inputs ----
    k_f32_to_f16<<<cdiv((long)N * F, 256), 256, 0, stream>>>(x, x16, (size_t)N * F);
    k_transpose_to_f16<<<cdiv(F * HC1, 256), 256, 0, stream>>>(W_l1, WlT1, F, HC1);
    k_transpose_to_f16<<<cdiv(F * HC1, 256), 256, 0, stream>>>(W_r1, WrT1, F, HC1);
    k_transpose_to_f16<<<cdiv(HC1 * C2, 256), 256, 0, stream>>>(W_l2, WlT2, HC1, C2);
    k_transpose_to_f16<<<cdiv(HC1 * C2, 256), 256, 0, stream>>>(W_r2, WrT2, HC1, C2);

    // ---- layer 1 GEMMs: xs1 = x@Wl1+bl1, xd1 = x@Wr1+br1 ----
    {
        long waves = (long)cdiv(N, 16) * (HC1 / 16);
        int blocks = cdiv(waves, 8);
        k_wmma_gemm_bias<<<blocks, 256, 0, stream>>>(x16, WlT1, b_l1, xs1, N, F, HC1);
        k_wmma_gemm_bias<<<blocks, 256, 0, stream>>>(x16, WrT1, b_r1, xd1, N, F, HC1);
    }

    // ---- layer 1 attention ----
    hipMemsetAsync(den1, 0, (size_t)N * 4 * 4, stream);
    hipMemsetAsync(out1, 0, (size_t)N * HC1 * 4, stream);
    k_fill_u32<<<cdiv((long)N * 4, 256), 256, 0, stream>>>(max1, NEG_MAX_ENC, (size_t)N * 4);
    {
        int eb = cdiv(Etot, 8);
        k_edge_score<4, 64><<<eb, 256, 0, stream>>>(xs1, xd1, att1, srcA, dstA,
                                                    E, Etot, score1, max1);
        k_edge_ex<4><<<cdiv((long)Etot * 4, 256), 256, 0, stream>>>(score1, dstA, E, Etot,
                                                                    max1, den1);
        k_edge_scatter<4, 64><<<eb, 256, 0, stream>>>(xs1, score1, den1, srcA, dstA,
                                                      E, Etot, out1);
    }
    k_bias_relu_f16<<<cdiv((long)N * HC1, 256), 256, 0, stream>>>(out1, bias1, h16,
                                                                  HC1, (size_t)N * HC1);

    // ---- layer 2 GEMMs: xs2 = h@Wl2+bl2, xd2 = h@Wr2+br2 ----
    {
        long waves = (long)cdiv(N, 16) * (C2 / 16);
        int blocks = cdiv(waves, 8);
        k_wmma_gemm_bias<<<blocks, 256, 0, stream>>>(h16, WlT2, b_l2, xs2, N, HC1, C2);
        k_wmma_gemm_bias<<<blocks, 256, 0, stream>>>(h16, WrT2, b_r2, xd2, N, HC1, C2);
    }

    // ---- layer 2 attention ----
    hipMemsetAsync(den2, 0, (size_t)N * 4, stream);
    hipMemsetAsync(out2, 0, (size_t)N * C2 * 4, stream);
    k_fill_u32<<<cdiv(N, 256), 256, 0, stream>>>(max2, NEG_MAX_ENC, (size_t)N);
    {
        int eb = cdiv(Etot, 8);
        k_edge_score<1, 64><<<eb, 256, 0, stream>>>(xs2, xd2, att2, srcA, dstA,
                                                    E, Etot, score2, max2);
        k_edge_ex<1><<<cdiv(Etot, 256), 256, 0, stream>>>(score2, dstA, E, Etot,
                                                          max2, den2);
        k_edge_scatter<1, 64><<<eb, 256, 0, stream>>>(xs2, score2, den2, srcA, dstA,
                                                      E, Etot, out2);
    }

    // ---- mean pool (+bias2) and final linear ----
    hipMemsetAsync(sums, 0, (size_t)64 * C2 * 4, stream);
    hipMemsetAsync(counts, 0, (size_t)64 * 4, stream);
    k_pool_accum<<<cdiv((long)N * C2, 256), 256, 0, stream>>>(out2, bias2, batch,
                                                              sums, counts, N, C2);
    k_final_lin<<<64, 64, 0, stream>>>(sums, counts, W_lin, b_lin, out, C2);
}
